// SubjectMLPLayers_26852135534914
// MI455X (gfx1250) — matrix operand order repaired and checked by reading, not compiled, not run on hardware
//
#include <hip/hip_runtime.h>
#include <math.h>

// ---------------------------------------------------------------------------
// SubjectMLP fused kernel for gfx1250 (MI455X):
//   out[b] = gelu(x[b] @ w1[sid[b]]) @ w2[sid[b]] + bias[sid[b]]
// B=32, T=512, D=512, MID=1024, O=512, all fp32 in memory.
//
// bf16 WMMA (v_wmma_f32_16x16x32_bf16) with fp32 accumulation, computed in
// TRANSPOSED form:  h^T = w1^T x^T  and  out^T = w2^T h^T.  With this
// orientation every LDS fragment read is a contiguous 16B ds_load_b128,
// the h^T D-tile store is one packed ds_store_b128, and the out^T epilogue
// is two contiguous global_store_b128 per tile (lane = token column).
// f32->bf16 uses native __bf16 casts + packed vector stores so the backend
// can emit packed cvt ops instead of 3-instruction integer rounding.
// ---------------------------------------------------------------------------

typedef __attribute__((ext_vector_type(16))) __bf16        v16bf;
typedef __attribute__((ext_vector_type(8)))  __bf16        bf16x8;
typedef __attribute__((ext_vector_type(4)))  __bf16        bf16x4;
typedef __attribute__((ext_vector_type(8)))  float         v8f;
typedef __attribute__((ext_vector_type(4)))  unsigned int  u32x4;
typedef __attribute__((ext_vector_type(4)))  float         f32x4;

union FragBF {
    v16bf bf;
    u32x4 u4[2];
};

__device__ __forceinline__ float gelu_exact(float v) {
    return 0.5f * v * (1.0f + erff(v * 0.70710678118654752f));
}

#define TM        32          // token rows per workgroup
#define DDIM      512
#define MIDDIM    1024
#define ODIM      512
#define XS_STRIDE 520         // D + 8 pad (bf16 elems) -> 1040 B (16B multiple)
#define HS_STRIDE 1032        // MID + 8 pad            -> 2064 B (16B multiple)
#define W_STRIDE  40          // 32 K elems + 8 pad     ->   80 B (16B multiple)
#define SMEM_USHORTS (TM * XS_STRIDE + TM * HS_STRIDE + MIDDIM * W_STRIDE)
#define SMEM_BYTES   (SMEM_USHORTS * 2)   // 181,248 bytes

__global__ __launch_bounds__(256, 1)
void subject_mlp_wmma(const float* __restrict__ xg,
                      const int*   __restrict__ sids,
                      const float* __restrict__ w1g,
                      const float* __restrict__ w2g,
                      const float* __restrict__ bg,
                      float*       __restrict__ outg) {
    const int tid  = threadIdx.x;
    const int wave = tid >> 5;
    const int lane = tid & 31;
    const int lo16 = lane & 15;
    const int hi   = lane >> 4;   // 0 or 1 (lane half)

    const int batch = blockIdx.x >> 4;           // 32 batches
    const int r0    = (blockIdx.x & 15) * TM;    // token-row block [r0, r0+32)

    const int sid = sids[batch];
    const float* __restrict__ w1s = w1g + (size_t)sid * (DDIM * MIDDIM);
    const float* __restrict__ w2s = w2g + (size_t)sid * (MIDDIM * ODIM);
    const float* __restrict__ bs  = bg  + (size_t)sid * ODIM;

    extern __shared__ __bf16 smem[];
    __bf16* xs   = smem;                        // TM   x XS_STRIDE  [token][d]
    __bf16* hs   = xs + TM * XS_STRIDE;         // TM   x HS_STRIDE  [token][mid]
    __bf16* wbuf = hs + TM * HS_STRIDE;         // 1024 x W_STRIDE   [n][kk] (transposed slice)

    // ---- stage x block: TM x 512 fp32 -> bf16 LDS, row-major, packed ----
    {
        const float* xrow = xg + ((size_t)batch * 512 + r0) * DDIM;
#pragma unroll
        for (int t = 0; t < 16; ++t) {
            int idx = tid + t * 256;          // [0, 4096) float4s
            int row = idx >> 7;               // 128 float4 per row
            int c4  = (idx & 127) << 2;
            f32x4 v = *(const f32x4*)(xrow + (size_t)row * DDIM + c4);
            bf16x4 pk;
            pk.x = (__bf16)v.x; pk.y = (__bf16)v.y;
            pk.z = (__bf16)v.z; pk.w = (__bf16)v.w;
            *(bf16x4*)(xs + row * XS_STRIDE + c4) = pk;
        }
    }

    // =====================================================================
    // Stage 1:  h^T = w1^T @ x^T   (M = MID, N = tokens)
    // wave owns 8 M-tiles (128 mid cols) x 2 token tiles -> 16 accumulators
    // =====================================================================
    v8f acc1[8][2];
#pragma unroll
    for (int mt = 0; mt < 8; ++mt)
#pragma unroll
        for (int tt = 0; tt < 2; ++tt) {
            v8f z = {0.f, 0.f, 0.f, 0.f, 0.f, 0.f, 0.f, 0.f};
            acc1[mt][tt] = z;
        }

    for (int ks = 0; ks < DDIM / 32; ++ks) {
        // stage w1 K-slice transposed: wbuf[m][kk].  Column loads are
        // coalesced across lanes (consecutive m); store is one packed b64.
#pragma unroll
        for (int t = 0; t < 32; ++t) {
            int idx = tid + t * 256;          // [0, 8192) groups of 4
            int m   = idx & 1023;
            int kk4 = (idx >> 10) << 2;       // 0,4,...,28
            const float* src = w1s + (size_t)(ks * 32 + kk4) * MIDDIM + m;
            bf16x4 pk;
            pk.x = (__bf16)src[0 * MIDDIM];
            pk.y = (__bf16)src[1 * MIDDIM];
            pk.z = (__bf16)src[2 * MIDDIM];
            pk.w = (__bf16)src[3 * MIDDIM];
            *(bf16x4*)(wbuf + m * W_STRIDE + kk4) = pk;
        }
        if (ks + 1 < DDIM / 32) {
            const float* nxt = w1s + (size_t)(ks + 1) * 32 * MIDDIM;
            __builtin_prefetch(nxt + (size_t)tid * 128, 0, 2);
        }
        __syncthreads();

        // B fragments from x (lane = token column; K 0-15 / 16-31 contiguous)
        FragBF bfr[2];
#pragma unroll
        for (int tt = 0; tt < 2; ++tt) {
            const __bf16* p = xs + (tt * 16 + lo16) * XS_STRIDE + ks * 32 + hi * 16;
            bfr[tt].u4[0] = *(const u32x4*)p;
            bfr[tt].u4[1] = *(const u32x4*)(p + 8);
        }
#pragma unroll
        for (int mt = 0; mt < 8; ++mt) {
            // A fragment from w1^T slice (lane = mid row; K 0-15 / 16-31 contiguous)
            FragBF a;
            const __bf16* p = wbuf + (wave * 128 + mt * 16 + lo16) * W_STRIDE + hi * 16;
            a.u4[0] = *(const u32x4*)p;
            a.u4[1] = *(const u32x4*)(p + 8);
#pragma unroll
            for (int tt = 0; tt < 2; ++tt)
                acc1[mt][tt] = __builtin_amdgcn_wmma_f32_16x16x32_bf16(
                    false, a.bf, false, bfr[tt].bf, (short)0, acc1[mt][tt],
                    false, false);
        }
        __syncthreads();
    }

    // GELU in registers; h^T D-tile (lane = token, r = consecutive mid)
    // packs to ONE ds_store_b128 per tile.
#pragma unroll
    for (int mt = 0; mt < 8; ++mt)
#pragma unroll
        for (int tt = 0; tt < 2; ++tt) {
            int token = tt * 16 + lo16;
            int mid0  = wave * 128 + mt * 16 + hi * 8;
            bf16x8 pk;
#pragma unroll
            for (int r = 0; r < 8; ++r)
                pk[r] = (__bf16)gelu_exact(acc1[mt][tt][r]);
            *(bf16x8*)(hs + token * HS_STRIDE + mid0) = pk;
        }
    __syncthreads();

    // =====================================================================
    // Stage 2:  out^T = w2^T @ h^T   (M = O, N = tokens)
    // wave owns 4 M-tiles (64 out cols) x 2 token tiles -> 8 accumulators
    // =====================================================================
    v8f acc2[4][2];
#pragma unroll
    for (int mt = 0; mt < 4; ++mt)
#pragma unroll
        for (int tt = 0; tt < 2; ++tt) {
            v8f z = {0.f, 0.f, 0.f, 0.f, 0.f, 0.f, 0.f, 0.f};
            acc2[mt][tt] = z;
        }

    for (int ks = 0; ks < MIDDIM / 32; ++ks) {
        // stage w2 K-slice transposed: wbuf[o][kk]
#pragma unroll
        for (int t = 0; t < 16; ++t) {
            int idx = tid + t * 256;          // [0, 4096) groups of 4
            int o   = idx & 511;
            int kk4 = (idx >> 9) << 2;        // 0,4,...,28
            const float* src = w2s + (size_t)(ks * 32 + kk4) * ODIM + o;
            bf16x4 pk;
            pk.x = (__bf16)src[0 * ODIM];
            pk.y = (__bf16)src[1 * ODIM];
            pk.z = (__bf16)src[2 * ODIM];
            pk.w = (__bf16)src[3 * ODIM];
            *(bf16x4*)(wbuf + o * W_STRIDE + kk4) = pk;
        }
        if (ks + 1 < MIDDIM / 32) {
            const float* nxt = w2s + (size_t)(ks + 1) * 32 * ODIM;
            __builtin_prefetch(nxt + (size_t)tid * 64, 0, 2);
        }
        __syncthreads();

        // B fragments from h (lane = token column; contiguous K)
        FragBF bfr[2];
#pragma unroll
        for (int tt = 0; tt < 2; ++tt) {
            const __bf16* p = hs + (tt * 16 + lo16) * HS_STRIDE + ks * 32 + hi * 16;
            bfr[tt].u4[0] = *(const u32x4*)p;
            bfr[tt].u4[1] = *(const u32x4*)(p + 8);
        }
#pragma unroll
        for (int mt = 0; mt < 4; ++mt) {
            FragBF a;
            const __bf16* p = wbuf + (wave * 64 + mt * 16 + lo16) * W_STRIDE + hi * 16;
            a.u4[0] = *(const u32x4*)p;
            a.u4[1] = *(const u32x4*)(p + 8);
#pragma unroll
            for (int tt = 0; tt < 2; ++tt)
                acc2[mt][tt] = __builtin_amdgcn_wmma_f32_16x16x32_bf16(
                    false, a.bf, false, bfr[tt].bf, (short)0, acc2[mt][tt],
                    false, false);
        }
        __syncthreads();
    }

    // Epilogue: out^T D-tile -> lane = token, r = 8 consecutive out columns.
    // Vectorized bias load + two contiguous 16B global stores per tile.
#pragma unroll
    for (int mt = 0; mt < 4; ++mt) {
        int o0 = wave * 64 + mt * 16 + hi * 8;     // 8 consecutive out columns
        f32x4 bv0 = *(const f32x4*)(bs + o0);
        f32x4 bv1 = *(const f32x4*)(bs + o0 + 4);
#pragma unroll
        for (int tt = 0; tt < 2; ++tt) {
            int token = r0 + tt * 16 + lo16;
            float* dst = outg + ((size_t)batch * 512 + token) * ODIM + o0;
            f32x4 s0, s1;
            s0.x = acc2[mt][tt][0] + bv0.x;
            s0.y = acc2[mt][tt][1] + bv0.y;
            s0.z = acc2[mt][tt][2] + bv0.z;
            s0.w = acc2[mt][tt][3] + bv0.w;
            s1.x = acc2[mt][tt][4] + bv1.x;
            s1.y = acc2[mt][tt][5] + bv1.y;
            s1.z = acc2[mt][tt][6] + bv1.z;
            s1.w = acc2[mt][tt][7] + bv1.w;
            *(f32x4*)dst       = s0;
            *(f32x4*)(dst + 4) = s1;
        }
    }
}

extern "C" void kernel_launch(void* const* d_in, const int* in_sizes, int n_in,
                              void* d_out, int out_size, void* d_ws, size_t ws_size,
                              hipStream_t stream) {
    (void)in_sizes; (void)n_in; (void)out_size; (void)d_ws; (void)ws_size;
    const float* x   = (const float*)d_in[0];
    const int*   sid = (const int*)  d_in[1];
    const float* w1  = (const float*)d_in[2];
    const float* w2  = (const float*)d_in[3];
    const float* b   = (const float*)d_in[4];
    float*       out = (float*)d_out;

    // Opt in to >64KB dynamic LDS (idempotent; deterministic every call).
    hipFuncSetAttribute((const void*)subject_mlp_wmma,
                        hipFuncAttributeMaxDynamicSharedMemorySize, SMEM_BYTES);

    dim3 grid(32 * (512 / TM));   // batch * row-blocks = 512 workgroups
    dim3 block(256);              // 8 wave32
    subject_mlp_wmma<<<grid, block, SMEM_BYTES, stream>>>(x, sid, w1, w2, b, out);
}
